// CorrelationLayer_85907935855279
// MI455X (gfx1250) — compile-verified, matrix operand order checked by reading, not run on hardware
//
#include <hip/hip_runtime.h>

// PWC-Net correlation / cost volume for MI455X (gfx1250, wave32).
// out[b, (dr+4)*9+(dc+4), h, w] = (1/128) * sum_c in0[b,c,h,w] * in1[b,c,h+dr,w+dc]
//
// Block = 8 waves = 8 consecutive output rows of one 16-pixel column tile.
// Stage in1[b, :, h0-4..h0+11, w0-4..w0+27] ONCE into LDS as bf16 hi/lo planes
// (3-term split product recovers ~fp32 accuracy), stored in WMMA B-fragment
// order [row][col][8-channel chunk] with XOR swizzle -> fragments are built
// with 2x ds_load_b128 per plane. 256KB staging + 16KB D tiles uses the CDNA5
// 320KB-per-WGP LDS. Zero padding applied at fill time only.
// The two 16-col N-tiles are computed with two INDEPENDENT accumulator chains,
// interleaved, so WMMA bursts hide ds_load latency instead of s_wait_dscnt 0x0
// draining before every matrix op.

#define MD     4
#define ND     9           // 2*MD+1
#define NDISP  81
#define C_CH   128
#define H_DIM  96
#define W_DIM  192
#define TILE   16
#define WTILES (W_DIM / TILE)

#define RPB        8                   // output rows per block (= waves per block)
#define SROWS      (RPB + 2 * MD)      // 16 staged in1 rows
#define SCOLS      32                  // staged in1 columns (24 needed, 32 for tiles)
#define NCHUNK     (C_CH / 8)          // 16 chunks of 8 channels

typedef __attribute__((ext_vector_type(8)))  __bf16 v8bf;
typedef __attribute__((ext_vector_type(16))) __bf16 v16bf;
typedef __attribute__((ext_vector_type(8)))  float  v8f;

__device__ __forceinline__ int iclamp(int v, int lo, int hi) {
    return v < lo ? lo : (v > hi ? hi : v);
}

__global__ __launch_bounds__(32 * RPB)
void corr_wmma_kernel(const float* __restrict__ in0,
                      const float* __restrict__ in1,
                      float* __restrict__ out)
{
    const int lane = threadIdx.x & 31;
    const int wid  = threadIdx.y;      // 0..7, wave id = output row within block
    const int half = lane >> 4;        // which 16-lane half
    const int lrow = lane & 15;        // M for A-frag, N for B-frag / D

    const int w0 = blockIdx.x * TILE;  // 0..176
    const int h0 = blockIdx.y * RPB;   // 0..88
    const int b  = blockIdx.z;         // 0..7
    const int h  = h0 + wid;           // this wave's output row

    const long HW = (long)H_DIM * W_DIM;

    // [plane][row][col][chunk][elem]: 2*16*32*16*8 bf16 = 256 KB
    __shared__ __align__(16) __bf16 sb[2][SROWS][SCOLS][NCHUNK][8];
    __shared__ float dtile[RPB][2][TILE][TILE];   // 16 KB, per-wave D staging

    // ---------------- cooperative fill: convert in1 slab to bf16 hi/lo ------
    {
        const float* src_b = in1 + (long)b * C_CH * HW;
        const int col = lane;                       // 0..31
        const int q   = w0 - MD + col;
        const bool colOK = (q >= 0) && (q < W_DIM);
        const int qc = iclamp(q, 0, W_DIM - 1);
        const int swz = col & 15;
#pragma unroll
        for (int rr = 0; rr < 2; ++rr) {
            const int s    = wid * 2 + rr;          // staged row 0..15
            const int grow = h0 - MD + s;
            const bool ok  = colOK && (grow >= 0) && (grow < H_DIM);
            const int gc   = iclamp(grow, 0, H_DIM - 1);
            const float* src = src_b + (long)gc * W_DIM + qc;
#pragma unroll
            for (int chg = 0; chg < NCHUNK; ++chg) {
                v8bf hi8, lo8;
#pragma unroll
                for (int e = 0; e < 8; ++e) {
                    float v = src[(long)(chg * 8 + e) * HW];
                    v = ok ? v : 0.0f;              // cndmask, no branch
                    __bf16 hv = (__bf16)v;
                    hi8[e] = hv;
                    lo8[e] = (__bf16)(v - (float)hv);
                }
                const int pch = chg ^ swz;          // bank swizzle
                *(v8bf*)&sb[0][s][col][pch][0] = hi8;
                *(v8bf*)&sb[1][s][col][pch][0] = lo8;
            }
        }
    }

    // ---------------- A fragments: in0[b, :, h, w0+lrow], hi/lo split -------
    // ISA 7.12.2: element i of the per-lane 16-vector maps to slab-K
    // = (i&8)*2 + 8*half + (i&7)  -> two contiguous 8-channel runs.
    v16bf a_hi[4], a_lo[4];
    {
        const float* p0 = in0 + (long)b * C_CH * HW + (long)h * W_DIM + (w0 + lrow);
#pragma unroll
        for (int kk = 0; kk < 4; ++kk) {
#pragma unroll
            for (int i = 0; i < 16; ++i) {
                const int kl = ((i & 8) << 1) + (half << 3) + (i & 7);
                float v = p0[(long)(kk * 32 + kl) * HW];
                __bf16 hv = (__bf16)v;
                a_hi[kk][i] = hv;
                a_lo[kk][i] = (__bf16)(v - (float)hv);
            }
        }
    }

    __syncthreads();   // staging complete; sb is read-only from here on

    const float scale = 1.0f / (float)C_CH;
    float* outbase = out + ((long)b * NDISP * H_DIM + h) * W_DIM + w0;

    const int col0 = lrow;            // N-tile 0 staged column
    const int col1 = lrow + 16;       // N-tile 1 staged column
    const int swz0 = col0 & 15;
    const int swz1 = col1 & 15;

#pragma unroll 1
    for (int r = 0; r < ND; ++r) {
        const int s = wid + r;                      // staged row for this shift

        v8f acc0 = {};
        v8f acc1 = {};
#pragma unroll
        for (int kk = 0; kk < 4; ++kk) {
            const int c0 = 4 * kk + half;           // chunk of ch 32kk+8*half
            const int c1 = c0 + 2;                  // chunk of ch 32kk+16+8*half

            // tile 0 fragments (hi, lo), each = two ds_load_b128
            v16bf bh_t0 = __builtin_shufflevector(
                    *(const v8bf*)&sb[0][s][col0][c0 ^ swz0][0],
                    *(const v8bf*)&sb[0][s][col0][c1 ^ swz0][0],
                    0,1,2,3,4,5,6,7,8,9,10,11,12,13,14,15);
            v16bf bl_t0 = __builtin_shufflevector(
                    *(const v8bf*)&sb[1][s][col0][c0 ^ swz0][0],
                    *(const v8bf*)&sb[1][s][col0][c1 ^ swz0][0],
                    0,1,2,3,4,5,6,7,8,9,10,11,12,13,14,15);
            // tile 1 fragments
            v16bf bh_t1 = __builtin_shufflevector(
                    *(const v8bf*)&sb[0][s][col1][c0 ^ swz1][0],
                    *(const v8bf*)&sb[0][s][col1][c1 ^ swz1][0],
                    0,1,2,3,4,5,6,7,8,9,10,11,12,13,14,15);
            v16bf bl_t1 = __builtin_shufflevector(
                    *(const v8bf*)&sb[1][s][col1][c0 ^ swz1][0],
                    *(const v8bf*)&sb[1][s][col1][c1 ^ swz1][0],
                    0,1,2,3,4,5,6,7,8,9,10,11,12,13,14,15);

            // Two independent accumulator chains, interleaved: no back-to-back
            // RAW between consecutive WMMAs, loads of next kk overlap the burst.
            acc0 = __builtin_amdgcn_wmma_f32_16x16x32_bf16(
                    false, a_lo[kk], false, bh_t0, (short)0, acc0, false, false);
            acc1 = __builtin_amdgcn_wmma_f32_16x16x32_bf16(
                    false, a_lo[kk], false, bh_t1, (short)0, acc1, false, false);
            acc0 = __builtin_amdgcn_wmma_f32_16x16x32_bf16(
                    false, a_hi[kk], false, bl_t0, (short)0, acc0, false, false);
            acc1 = __builtin_amdgcn_wmma_f32_16x16x32_bf16(
                    false, a_hi[kk], false, bl_t1, (short)0, acc1, false, false);
            acc0 = __builtin_amdgcn_wmma_f32_16x16x32_bf16(
                    false, a_hi[kk], false, bh_t0, (short)0, acc0, false, false);
            acc1 = __builtin_amdgcn_wmma_f32_16x16x32_bf16(
                    false, a_hi[kk], false, bh_t1, (short)0, acc1, false, false);
        }

        // D layout: VGPR j -> M = j + 8*half, N = lrow
#pragma unroll
        for (int j = 0; j < 8; ++j) {
            dtile[wid][0][j + (half << 3)][lrow] = acc0[j];
            dtile[wid][1][j + (half << 3)][lrow] = acc1[j];
        }

        // Banded extraction: out pixel m, col-shift cp -> D[m, m+cp] across the
        // 2 tiles. Per-wave private dtile region; in-wave DS ordering suffices.
#pragma unroll
        for (int it = 0; it < 5; ++it) {
            const int idx = lane + it * 32;
            if (idx < ND * TILE) {
                const int cp = idx >> 4;            // 0..8  (dc = cp-4)
                const int m  = idx & 15;            // output pixel within tile
                const int n  = m + cp;              // 0..23
                const float v = (n < TILE) ? dtile[wid][0][m][n]
                                           : dtile[wid][1][m][n - TILE];
                outbase[(long)(r * ND + cp) * (H_DIM * W_DIM) + m] = v * scale;
            }
        }
    }
}

extern "C" void kernel_launch(void* const* d_in, const int* in_sizes, int n_in,
                              void* d_out, int out_size, void* d_ws, size_t ws_size,
                              hipStream_t stream) {
    const float* in0 = (const float*)d_in[0];
    const float* in1 = (const float*)d_in[1];
    float* out = (float*)d_out;
    (void)in_sizes; (void)n_in; (void)out_size; (void)d_ws; (void)ws_size;

    dim3 grid(WTILES, H_DIM / RPB, 8);   // 12 x 12 x 8 = 1152 blocks
    dim3 block(32, RPB, 1);              // 8 waves (wave32)
    corr_wmma_kernel<<<grid, block, 0, stream>>>(in0, in1, out);
}
